// Head_69999376990240
// MI455X (gfx1250) — matrix operand order repaired and checked by reading, not compiled, not run on hardware
//
#include <hip/hip_runtime.h>
#include <hip/hip_bf16.h>

// Problem constants (B=4, T=4096, D=1024, K=64)
#define BT     16384    // B*T
#define DMODEL 1024
#define HK     64
#define TLEN   4096
#define LDX    1030     // padded LDS stride (elements) for x tile in kernel 1
#define PLD    36       // padded LDS stride (elements) for P tile
#define KLD    72       // K-tile LDS row stride (elements) = 144B, keeps 16B chunks aligned

typedef __attribute__((ext_vector_type(16))) __bf16 v16bf;
typedef __attribute__((ext_vector_type(8)))  float  v8f;

// ---- CDNA5 async global->LDS copy (ASYNCcnt) with portable fallback --------
#if defined(__has_builtin)
#if __has_builtin(__builtin_amdgcn_global_load_async_to_lds_b128) && \
    __has_builtin(__builtin_amdgcn_s_wait_asynccnt)
#define HAVE_ASYNC 1
#endif
#endif
#ifndef HAVE_ASYNC
#define HAVE_ASYNC 0
#endif

typedef __attribute__((vector_size(4 * sizeof(int)))) int v4i_t;
typedef __attribute__((address_space(1))) v4i_t gv4i;   // global b128 chunk
typedef __attribute__((address_space(3))) v4i_t lv4i;   // LDS    b128 chunk

static __device__ __forceinline__ void copy16_g2l(const __bf16* g, __bf16* l) {
#if HAVE_ASYNC
    __builtin_amdgcn_global_load_async_to_lds_b128(
        (gv4i*)(void*)g, (lv4i*)(void*)l, 0, 0);
#else
    *(float4*)l = *(const float4*)g;
#endif
}
static __device__ __forceinline__ void async_drain() {
#if HAVE_ASYNC
    __builtin_amdgcn_s_wait_asynccnt(0);
#endif
}

static __device__ __forceinline__ v8f wmma_bf16(v16bf a, v16bf b, v8f c) {
    return __builtin_amdgcn_wmma_f32_16x16x32_bf16(false, a, false, b, (short)0, c,
                                                   false, false);
}

// A-fragment K offset for 16-bit 16x32 A-matrix (contiguous order: k = 16*half + s)
static __device__ __forceinline__ int a_koff(int i, int half) {
    return half * 16 + 2 * i;
}

// ---------------------------------------------------------------------------
// Kernel 1: fused QKV projection, fp32 -> bf16 Q (pre-scaled), K, V (pair-
// interleaved row-major: element (s,c) at ((s>>1)*D + c)*2 + (s&1))
// ---------------------------------------------------------------------------
__global__ __launch_bounds__(256) void qkv_kernel(
    const float* __restrict__ x,
    const float* __restrict__ Wq,
    const float* __restrict__ Wk,
    const float* __restrict__ Wv,
    __bf16* __restrict__ Qb,     // (BT, 64) row-major
    __bf16* __restrict__ Kb,     // (BT, 64) row-major
    __bf16* __restrict__ V2)     // (BT/2, DMODEL, 2) pair-interleaved
{
    __shared__ __bf16 xs[16 * LDX];

    const int tid  = threadIdx.x;
    const int mt   = blockIdx.x;          // 16-row tile of x, 0..1023
    const int wave = tid >> 5;
    const int lane = tid & 31;
    const int lm   = lane & 15;
    const int half = lane >> 4;

    // Stage 16x1024 fp32 x-tile into LDS as bf16 (one row per iteration).
    for (int i = 0; i < 16; ++i) {
        const float4 f =
            *(const float4*)(x + (size_t)(mt * 16 + i) * DMODEL + tid * 4);
        __bf16* p = &xs[i * LDX + tid * 4];
        p[0] = (__bf16)f.x; p[1] = (__bf16)f.y;
        p[2] = (__bf16)f.z; p[3] = (__bf16)f.w;
    }
    __syncthreads();

    // 72 column tiles total: [0,4)=Q, [4,8)=K, [8,72)=V. 9 per wave.
    for (int j = 0; j < 9; ++j) {
        const int nt = wave * 9 + j;
        const float* W = (nt < 4) ? Wq : ((nt < 8) ? Wk : Wv);
        const int ldw  = (nt < 8) ? HK : DMODEL;
        const int col0 = (nt < 4) ? nt * 16
                                  : ((nt < 8) ? (nt - 4) * 16 : (nt - 8) * 16);

        v8f acc = {};
        for (int kc = 0; kc < DMODEL; kc += 32) {
            v16bf a;
#pragma unroll
            for (int i = 0; i < 8; ++i) {
                const int kb = kc + a_koff(i, half);
                a[2 * i]     = xs[lm * LDX + kb];
                a[2 * i + 1] = xs[lm * LDX + kb + 1];
            }
            v16bf bw;
#pragma unroll
            for (int v = 0; v < 8; ++v) {
                const int k0 = kc + half * 16 + 2 * v;
                bw[2 * v]     = (__bf16)W[(size_t)k0 * ldw + col0 + lm];
                bw[2 * v + 1] = (__bf16)W[(size_t)(k0 + 1) * ldw + col0 + lm];
            }
            acc = wmma_bf16(a, bw, acc);
        }

        const int gm0 = mt * 16;
        if (nt < 8) {
            __bf16* out   = (nt < 4) ? Qb : Kb;
            const float s = (nt < 4) ? 0.125f : 1.0f;   // fold 64^-0.5 into Q
#pragma unroll
            for (int r = 0; r < 8; ++r) {
                const int m = r + 8 * half;
                out[(size_t)(gm0 + m) * HK + col0 + lm] = (__bf16)(acc[r] * s);
            }
        } else {
#pragma unroll
            for (int r = 0; r < 8; ++r) {
                const int gm = gm0 + r + 8 * half;
                V2[((size_t)(gm >> 1) * DMODEL + col0 + lm) * 2 + (gm & 1)] =
                    (__bf16)acc[r];
            }
        }
    }
}

// ---------------------------------------------------------------------------
// Kernel 2: causal flash attention. 1 block = 16 queries; wave w owns output
// columns [w*128, w*128+128). Streams 32-key steps up to the diagonal with
// K tiles double-buffered in LDS via async global->LDS copies.
// ---------------------------------------------------------------------------
__global__ __launch_bounds__(256) void attn_kernel(
    const __bf16* __restrict__ Qb,
    const __bf16* __restrict__ Kb,
    const __bf16* __restrict__ V2,
    float* __restrict__ out)
{
    __shared__ __bf16 ktile[2][32 * KLD];    // double-buffered 32x64 K tile
    __shared__ __bf16 pshare[8][16 * PLD];   // per-wave P re-layout staging

    const int tid  = threadIdx.x;
    const int wave = tid >> 5;
    const int lane = tid & 31;
    const int lm   = lane & 15;
    const int half = lane >> 4;
    const int krow   = tid >> 3;             // 0..31 : K-tile row this thread stages
    const int kchunk = tid & 7;              // 8 x 16B chunks per 128B row

    const int bid   = blockIdx.x;
    const int bb    = bid >> 8;              // batch (T/16 = 256 tiles each)
    const int qt    = bid & 255;
    const int qbase = qt * 16;
    const size_t rowbase = (size_t)bb * TLEN;

    // Q tile as two 16x32 A fragments (K = 0..31 and 32..63)
    v16bf aq0, aq1;
    {
        const __bf16* qrow = Qb + (rowbase + qbase + lm) * HK;
#pragma unroll
        for (int i = 0; i < 8; ++i) {
            const int kb = a_koff(i, half);
            aq0[2 * i]     = qrow[kb];
            aq0[2 * i + 1] = qrow[kb + 1];
            aq1[2 * i]     = qrow[32 + kb];
            aq1[2 * i + 1] = qrow[32 + kb + 1];
        }
    }

    v8f acc[8];
    v8f zero = {};
#pragma unroll
    for (int i = 0; i < 8; ++i) acc[i] = zero;

    float mrow[8], lrow[8];
#pragma unroll
    for (int r = 0; r < 8; ++r) { mrow[r] = -1e30f; lrow[r] = 0.0f; }

    const int nsteps = (qbase >> 5) + 1;     // causal: only up to the diagonal
    const int c0w    = wave * 128;
    __bf16* pw       = &pshare[wave][0];

    // Stage K tile for step 0 (one b128 per thread)
    copy16_g2l(Kb + (rowbase + krow) * HK + kchunk * 8,
               &ktile[0][krow * KLD + kchunk * 8]);

    for (int st = 0; st < nsteps; ++st) {
        const int s0 = st * 32;

        // Publish current K buffer; everyone is also done with the other one.
        async_drain();
        __syncthreads();
        if (st + 1 < nsteps) {
            copy16_g2l(Kb + (rowbase + (st + 1) * 32 + krow) * HK + kchunk * 8,
                       &ktile[(st + 1) & 1][krow * KLD + kchunk * 8]);
        }

        // K fragments from LDS: 2 key tiles x 2 contraction chunks
        const __bf16* kt = &ktile[st & 1][0];
        v16bf bk00, bk01, bk10, bk11;
#pragma unroll
        for (int v = 0; v < 8; ++v) {
            const int kk = half * 16 + 2 * v;
            const __bf16* r0 = kt + lm * KLD;
            const __bf16* r1 = kt + (16 + lm) * KLD;
            bk00[2 * v] = r0[kk];      bk00[2 * v + 1] = r0[kk + 1];
            bk01[2 * v] = r0[32 + kk]; bk01[2 * v + 1] = r0[32 + kk + 1];
            bk10[2 * v] = r1[kk];      bk10[2 * v + 1] = r1[kk + 1];
            bk11[2 * v] = r1[32 + kk]; bk11[2 * v + 1] = r1[32 + kk + 1];
        }

        // S = Q * K^T  (16 x 32 scores, scale pre-folded into Q)
        v8f sf0 = zero, sf1 = zero;
        sf0 = wmma_bf16(aq0, bk00, sf0);
        sf0 = wmma_bf16(aq1, bk01, sf0);
        sf1 = wmma_bf16(aq0, bk10, sf1);
        sf1 = wmma_bf16(aq1, bk11, sf1);

        // Causal mask on the diagonal step (branchless per-lane selects)
        if (st == nsteps - 1) {
#pragma unroll
            for (int r = 0; r < 8; ++r) {
                const int q = qbase + r + 8 * half;
                sf0[r] = (s0 + lm      > q) ? -1e30f : sf0[r];
                sf1[r] = (s0 + 16 + lm > q) ? -1e30f : sf1[r];
            }
        }

        // Online softmax; key axis is the 16-lane dimension of each half
#pragma unroll
        for (int r = 0; r < 8; ++r) {
            float v = fmaxf(sf0[r], sf1[r]);
            v = fmaxf(v, __shfl_xor(v, 1, 32));
            v = fmaxf(v, __shfl_xor(v, 2, 32));
            v = fmaxf(v, __shfl_xor(v, 4, 32));
            v = fmaxf(v, __shfl_xor(v, 8, 32));
            const float mnew = fmaxf(mrow[r], v);
            const float corr = __expf(mrow[r] - mnew);
            mrow[r] = mnew;
            const float p0 = __expf(sf0[r] - mnew);
            const float p1 = __expf(sf1[r] - mnew);
            sf0[r] = p0; sf1[r] = p1;
            float ps = p0 + p1;
            ps += __shfl_xor(ps, 1, 32);
            ps += __shfl_xor(ps, 2, 32);
            ps += __shfl_xor(ps, 4, 32);
            ps += __shfl_xor(ps, 8, 32);
            lrow[r] = lrow[r] * corr + ps;
#pragma unroll
            for (int ct = 0; ct < 8; ++ct) acc[ct][r] *= corr;
        }

        // Re-layout P (C-format f32) -> A-format bf16 through per-wave LDS
#pragma unroll
        for (int r = 0; r < 8; ++r) {
            const int m = r + 8 * half;
            pw[m * PLD + lm]      = (__bf16)sf0[r];
            pw[m * PLD + 16 + lm] = (__bf16)sf1[r];
        }
        v16bf ap;
#pragma unroll
        for (int i = 0; i < 8; ++i) {
            const int kb = a_koff(i, half);
            ap[2 * i]     = pw[lm * PLD + kb];
            ap[2 * i + 1] = pw[lm * PLD + kb + 1];
        }

        // O += P * V ; pair-interleaved V: key pair (2v,2v+1) is one dword,
        // lanes read contiguous columns -> coalesced 64B segments.
        const size_t vsuper = (rowbase + s0) >> 1;
        __builtin_prefetch(V2 + ((vsuper + 16) * DMODEL + c0w + lm) * 2, 0, 1);
#pragma unroll
        for (int ct = 0; ct < 8; ++ct) {
            const int c = c0w + ct * 16 + lm;
            v16bf bv;
#pragma unroll
            for (int v = 0; v < 8; ++v) {
                const __bf16* p =
                    V2 + ((vsuper + half * 8 + v) * DMODEL + c) * 2;
                bv[2 * v]     = p[0];
                bv[2 * v + 1] = p[1];
            }
            acc[ct] = wmma_bf16(ap, bv, acc[ct]);
        }
    }

    // Epilogue: divide by running sum, write fp32
#pragma unroll
    for (int r = 0; r < 8; ++r) {
        const float inv = 1.0f / lrow[r];
        const int m = r + 8 * half;
        float* orow = out + (rowbase + qbase + m) * DMODEL + c0w + lm;
#pragma unroll
        for (int ct = 0; ct < 8; ++ct) orow[ct * 16] = acc[ct][r] * inv;
    }
}

// ---------------------------------------------------------------------------
extern "C" void kernel_launch(void* const* d_in, const int* in_sizes, int n_in,
                              void* d_out, int out_size, void* d_ws,
                              size_t ws_size, hipStream_t stream)
{
    (void)in_sizes; (void)n_in; (void)out_size; (void)ws_size;
    const float* x  = (const float*)d_in[0];
    const float* Wq = (const float*)d_in[1];
    const float* Wk = (const float*)d_in[2];
    const float* Wv = (const float*)d_in[3];

    // Workspace layout (36 MB total):
    //   Qb : (BT,64) bf16            =  2 MB
    //   Kb : (BT,64) bf16            =  2 MB
    //   V2 : (BT/2,DMODEL,2) bf16    = 32 MB
    unsigned char* ws = (unsigned char*)d_ws;
    __bf16* Qb = (__bf16*)(ws);
    __bf16* Kb = (__bf16*)(ws + (size_t)BT * HK * 2);
    __bf16* V2 = (__bf16*)(ws + (size_t)BT * HK * 4);

    qkv_kernel<<<BT / 16, 256, 0, stream>>>(x, Wq, Wk, Wv, Qb, Kb, V2);
    attn_kernel<<<BT / 16, 256, 0, stream>>>(Qb, Kb, V2, (float*)d_out);
}
